// MultiHeadAttention_1030792151059
// MI455X (gfx1250) — matrix operand order repaired and checked by reading, not compiled, not run on hardware
//
#include <hip/hip_runtime.h>
#include <stdint.h>

// ---------------------------------------------------------------------------
// MI455X / gfx1250 multi-head attention forward.
// M=2, T=2048, D=1024, H=16, HD=64.  All matrix math via v_wmma_f32_16x16x32_bf16.
// ---------------------------------------------------------------------------

typedef __attribute__((ext_vector_type(16))) __bf16 v16bf;
typedef __attribute__((ext_vector_type(8)))  float  v8f;
typedef __attribute__((ext_vector_type(4)))  int    v4i;

#define AS1 __attribute__((address_space(1)))
#define AS3 __attribute__((address_space(3)))

union FragAB { v16bf v; uint4 q[2]; };
union U4S8   { uint4 u; uint16_t s[8]; };

// Optional CDNA5 async global->LDS copy path (guarded; falls back cleanly).
#if defined(__has_builtin)
#if __has_builtin(__builtin_amdgcn_global_load_async_to_lds_b128) && \
    __has_builtin(__builtin_amdgcn_s_wait_asynccnt)
#define USE_ASYNC_LDS 1
#endif
#endif
#ifndef USE_ASYNC_LDS
#define USE_ASYNC_LDS 0
#endif

__device__ __forceinline__ uint16_t f2bf(float f) {
    uint32_t u = __builtin_bit_cast(uint32_t, f);
    u += 0x7FFFu + ((u >> 16) & 1u);          // round to nearest even
    return (uint16_t)(u >> 16);
}
__device__ __forceinline__ uint32_t pack2bf(float a, float b) {
    return (uint32_t)f2bf(a) | ((uint32_t)f2bf(b) << 16);
}
__device__ __forceinline__ uint4 pack8bf(const float4& a, const float4& b) {
    uint4 u;
    u.x = pack2bf(a.x, a.y); u.y = pack2bf(a.z, a.w);
    u.z = pack2bf(b.x, b.y); u.w = pack2bf(b.z, b.w);
    return u;
}

// A-matrix fragment (16x32 bf16): lane = row, dwords interleave K by 8*half.
// ISA 7.12.2: V0..3 lanes0-15 K=0..7, lanes16-31 K=8..15; V4..7 K=+16.
__device__ __forceinline__ FragAB load_frag_A(const uint16_t* rowp, int half) {
    FragAB f;
    f.q[0] = *(const uint4*)(rowp + 8 * half);
    f.q[1] = *(const uint4*)(rowp + 16 + 8 * half);
    return f;
}
// B-matrix fragment (32x16 bf16): lane = column, same K interleave as A (8*half, +16).
__device__ __forceinline__ FragAB load_frag_B(const uint16_t* colp, int half) {
    FragAB f;
    f.q[0] = *(const uint4*)(colp + 8 * half);
    f.q[1] = *(const uint4*)(colp + 16 + 8 * half);
    return f;
}

__device__ __forceinline__ float half_rmax(float v) {   // reduce across 16-lane half
    v = fmaxf(v, __shfl_xor(v, 1, 32));
    v = fmaxf(v, __shfl_xor(v, 2, 32));
    v = fmaxf(v, __shfl_xor(v, 4, 32));
    v = fmaxf(v, __shfl_xor(v, 8, 32));
    return v;
}
__device__ __forceinline__ float half_rsum(float v) {
    v += __shfl_xor(v, 1, 32);
    v += __shfl_xor(v, 2, 32);
    v += __shfl_xor(v, 4, 32);
    v += __shfl_xor(v, 8, 32);
    return v;
}

// ---------------------------------------------------------------------------
// Tiled GEMM:  out[Mrows x N] = A[Mrows x K] @ W[N x K]^T + bias  (nn.Linear)
// Macro tile 128x128, K-step 64 (2 WMMA sub-steps), 8 waves (2M x 4N) = 64x32/wave.
// ---------------------------------------------------------------------------
#define GT   128
#define GK   64
#define GLS  72   // LDS row stride (bf16 elems); 72*2=144B, 16B aligned

template <bool A_F32, bool OUT_F32>
__global__ __launch_bounds__(256)
void gemm_bf16_wmma(const void* __restrict__ Ap, const float* __restrict__ W,
                    const float* __restrict__ bias, void* __restrict__ outp,
                    int Kdim) {
    __shared__ uint16_t lA[GT * GLS];
    __shared__ uint16_t lB[GT * GLS];

    const int tid    = threadIdx.x;
    const int lane   = tid & 31;
    const int wave   = tid >> 5;
    const int lane16 = lane & 15;
    const int half   = lane >> 4;
    const int wm     = wave >> 2;   // 0..1
    const int wn     = wave & 3;    // 0..3
    const int mbase  = blockIdx.y * GT;
    const int nbase  = blockIdx.x * GT;
    const int N      = gridDim.x * GT;

    v8f acc[4][2];
#pragma unroll
    for (int rt = 0; rt < 4; ++rt)
#pragma unroll
        for (int ct = 0; ct < 2; ++ct) acc[rt][ct] = (v8f)(0.0f);

    for (int k0 = 0; k0 < Kdim; k0 += GK) {
        __syncthreads();
        // stage A and W tiles into LDS as bf16: 1024 chunks of 8 elems each
#pragma unroll
        for (int i = 0; i < 4; ++i) {
            int idx = tid + i * 256;
            int row = idx >> 3;
            int kc  = (idx & 7) * 8;
            if constexpr (A_F32) {
                const float* src = (const float*)Ap + (size_t)(mbase + row) * Kdim + k0 + kc;
                float4 f0 = *(const float4*)src;
                float4 f1 = *(const float4*)(src + 4);
                *(uint4*)&lA[row * GLS + kc] = pack8bf(f0, f1);
                if (k0 + GK < Kdim) __builtin_prefetch(src + GK, 0, 1);
            } else {
                const uint16_t* src = (const uint16_t*)Ap + (size_t)(mbase + row) * Kdim + k0 + kc;
#if USE_ASYNC_LDS
                __builtin_amdgcn_global_load_async_to_lds_b128(
                    (AS1 v4i*)(void*)const_cast<uint16_t*>(src),
                    (AS3 v4i*)(void*)&lA[row * GLS + kc], 0, 0);
#else
                *(uint4*)&lA[row * GLS + kc] = *(const uint4*)src;
#endif
                if (k0 + GK < Kdim) __builtin_prefetch(src + GK, 0, 1);
            }
            const float* wsrc = W + (size_t)(nbase + row) * Kdim + k0 + kc;
            float4 g0 = *(const float4*)wsrc;
            float4 g1 = *(const float4*)(wsrc + 4);
            *(uint4*)&lB[row * GLS + kc] = pack8bf(g0, g1);
            if (k0 + GK < Kdim) __builtin_prefetch(wsrc + GK, 0, 1);
        }
#if USE_ASYNC_LDS
        if constexpr (!A_F32) __builtin_amdgcn_s_wait_asynccnt(0);
#endif
        __syncthreads();

#pragma unroll
        for (int kk = 0; kk < 2; ++kk) {    // two 32-wide WMMA sub-steps
            FragAB a[4], b[2];
#pragma unroll
            for (int rt = 0; rt < 4; ++rt)
                a[rt] = load_frag_A(&lA[(wm * 64 + rt * 16 + lane16) * GLS + kk * 32], half);
#pragma unroll
            for (int ct = 0; ct < 2; ++ct)
                b[ct] = load_frag_B(&lB[(wn * 32 + ct * 16 + lane16) * GLS + kk * 32], half);

#pragma unroll
            for (int rt = 0; rt < 4; ++rt)
#pragma unroll
                for (int ct = 0; ct < 2; ++ct)
                    acc[rt][ct] = __builtin_amdgcn_wmma_f32_16x16x32_bf16(
                        false, a[rt].v, false, b[ct].v, (short)0, acc[rt][ct], false, false);
        }
    }

    // epilogue: + bias, store (C layout: vgpr r -> row r+8*half, lane16 -> col)
#pragma unroll
    for (int ct = 0; ct < 2; ++ct) {
        int gcol = nbase + wn * 32 + ct * 16 + lane16;
        float bv = bias[gcol];
#pragma unroll
        for (int rt = 0; rt < 4; ++rt) {
#pragma unroll
            for (int r = 0; r < 8; ++r) {
                int grow  = mbase + wm * 64 + rt * 16 + r + 8 * half;
                float val = acc[rt][ct][r] + bv;
                if constexpr (OUT_F32)
                    ((float*)outp)[(size_t)grow * N + gcol] = val;
                else
                    ((uint16_t*)outp)[(size_t)grow * N + gcol] = f2bf(val);
            }
        }
    }
}

// ---------------------------------------------------------------------------
// Flash-style attention. Block: 128 query rows for one (m,h); 8 waves x 16 rows.
// Inner loop over 64-key blocks: S via WMMA (K frags straight from global),
// online softmax in f32, P restaged via LDS into A-layout, O += P@V via WMMA.
// ---------------------------------------------------------------------------
#define AT   2048
#define AD   1024
#define AHD  64
#define ABQ  128
#define ABK  64
#define VSTR 72
#define PSTR 72

__global__ __launch_bounds__(256)
void attn_wmma(const uint16_t* __restrict__ qb, const uint16_t* __restrict__ kb,
               const uint16_t* __restrict__ vb, const int* __restrict__ mask,
               uint16_t* __restrict__ concat) {
    __shared__ uint16_t ldsV[AHD * VSTR];        // [hd][t_k] (transposed V tile)
    __shared__ uint16_t ldsP[8 * 16 * PSTR];     // per-wave 16x64 P tile

    const int tid    = threadIdx.x;
    const int lane   = tid & 31;
    const int wave   = tid >> 5;
    const int lane16 = lane & 15;
    const int half   = lane >> 4;
    const int qblk   = blockIdx.x;
    const int h      = blockIdx.y;
    const int m      = blockIdx.z;
    const float SCALE = 0.03125f;                // 1/sqrt(1024)

    // Q fragments for this wave's 16 rows, held in registers for entire loop.
    FragAB qa[2];
    {
        int row = qblk * ABQ + wave * 16 + lane16;
        const uint16_t* base = qb + ((size_t)(m * AT + row)) * AD + h * AHD;
#pragma unroll
        for (int kf = 0; kf < 2; ++kf) qa[kf] = load_frag_A(base + kf * 32, half);
    }

    v8f o[4];
#pragma unroll
    for (int nt = 0; nt < 4; ++nt) o[nt] = (v8f)(0.0f);
    float mrow[8], lrow[8];
#pragma unroll
    for (int r = 0; r < 8; ++r) { mrow[r] = -1e30f; lrow[r] = 0.0f; }

    uint16_t* myP = &ldsP[wave * 16 * PSTR];

    for (int kb0 = 0; kb0 < AT; kb0 += ABK) {
        __syncthreads();    // previous iteration's LDS reads done

        // prefetch next key block (K and V rows) while this one is computed
        if (kb0 + ABK < AT) {
            int t   = tid & 63;
            int sg  = (tid >> 6) * 16;
            size_t nrow = (size_t)(m * AT + kb0 + ABK + t) * AD + h * AHD + sg;
            __builtin_prefetch(kb + nrow, 0, 1);
            __builtin_prefetch(vb + nrow, 0, 1);
        }

        // stage V tile transposed: ldsV[hd][t] = v[kb0+t][hd]
#pragma unroll
        for (int i = 0; i < 2; ++i) {
            int idx = tid + i * 256;             // 512 chunks of 8 elems
            int t   = idx >> 3;
            int hd0 = (idx & 7) * 8;
            U4S8 tmp;
            tmp.u = *(const uint4*)(vb + ((size_t)(m * AT + kb0 + t)) * AD + h * AHD + hd0);
#pragma unroll
            for (int j = 0; j < 8; ++j) ldsV[(hd0 + j) * VSTR + t] = tmp.s[j];
        }

        // S = q @ k^T  (K fragments loaded directly from global, B layout)
        v8f s[4];
#pragma unroll
        for (int nt = 0; nt < 4; ++nt) {
            s[nt] = (v8f)(0.0f);
            int krow = kb0 + nt * 16 + lane16;   // key index = B column
            const uint16_t* base = kb + ((size_t)(m * AT + krow)) * AD + h * AHD;
#pragma unroll
            for (int ks = 0; ks < 2; ++ks) {
                FragAB bf = load_frag_B(base + ks * 32, half);
                s[nt] = __builtin_amdgcn_wmma_f32_16x16x32_bf16(
                    false, qa[ks].v, false, bf.v, (short)0, s[nt], false, false);
            }
        }

        // scale + mask
#pragma unroll
        for (int nt = 0; nt < 4; ++nt) {
            int mk = mask[m * AT + kb0 + nt * 16 + lane16];
#pragma unroll
            for (int r = 0; r < 8; ++r)
                s[nt][r] = (mk != 0) ? s[nt][r] * SCALE : -1e20f;
        }

        // online softmax (rows r+8*half live across 16 lanes of this half)
        float alpha[8];
#pragma unroll
        for (int r = 0; r < 8; ++r) {
            float rm = fmaxf(fmaxf(s[0][r], s[1][r]), fmaxf(s[2][r], s[3][r]));
            rm = half_rmax(rm);
            float mn = fmaxf(mrow[r], rm);
            alpha[r] = __expf(mrow[r] - mn);
            mrow[r]  = mn;
        }
#pragma unroll
        for (int r = 0; r < 8; ++r) {
            float rs = 0.0f;
#pragma unroll
            for (int nt = 0; nt < 4; ++nt) {
                float p = __expf(s[nt][r] - mrow[r]);
                s[nt][r] = p;
                rs += p;
            }
            rs = half_rsum(rs);
            lrow[r] = lrow[r] * alpha[r] + rs;
        }
#pragma unroll
        for (int nt = 0; nt < 4; ++nt)
#pragma unroll
            for (int r = 0; r < 8; ++r) o[nt][r] *= alpha[r];

        // P: C-layout -> LDS -> A-layout fragments
#pragma unroll
        for (int nt = 0; nt < 4; ++nt)
#pragma unroll
            for (int r = 0; r < 8; ++r)
                myP[(r + 8 * half) * PSTR + nt * 16 + lane16] = f2bf(s[nt][r]);
        __syncthreads();    // ldsV writes + own-wave P visible

        FragAB pf[2];
#pragma unroll
        for (int kf = 0; kf < 2; ++kf)
            pf[kf] = load_frag_A(&myP[lane16 * PSTR + kf * 32], half);

        // O += P @ V
#pragma unroll
        for (int nt = 0; nt < 4; ++nt) {
#pragma unroll
            for (int ks = 0; ks < 2; ++ks) {
                FragAB vf = load_frag_B(&ldsV[(nt * 16 + lane16) * VSTR + ks * 32], half);
                o[nt] = __builtin_amdgcn_wmma_f32_16x16x32_bf16(
                    false, pf[ks].v, false, vf.v, (short)0, o[nt], false, false);
            }
        }
    }

    // finalize: O /= l, write concat[m][t][h*64 + hd] as bf16
#pragma unroll
    for (int r = 0; r < 8; ++r) lrow[r] = 1.0f / lrow[r];
#pragma unroll
    for (int nt = 0; nt < 4; ++nt) {
        int gcol = h * AHD + nt * 16 + lane16;
#pragma unroll
        for (int r = 0; r < 8; ++r) {
            int grow = qblk * ABQ + wave * 16 + r + 8 * half;
            concat[((size_t)(m * AT + grow)) * AD + gcol] = f2bf(o[nt][r] * lrow[r]);
        }
    }
}

// ---------------------------------------------------------------------------
extern "C" void kernel_launch(void* const* d_in, const int* in_sizes, int n_in,
                              void* d_out, int out_size, void* d_ws, size_t ws_size,
                              hipStream_t stream) {
    const float* Q  = (const float*)d_in[0];
    const float* Kp = (const float*)d_in[1];
    const float* V  = (const float*)d_in[2];
    const int* mask = (const int*)d_in[3];
    const float* Wq = (const float*)d_in[4];  const float* bq = (const float*)d_in[5];
    const float* Wk = (const float*)d_in[6];  const float* bk = (const float*)d_in[7];
    const float* Wv = (const float*)d_in[8];  const float* bv = (const float*)d_in[9];
    const float* Wo = (const float*)d_in[10]; const float* bo = (const float*)d_in[11];
    float* out = (float*)d_out;

    const size_t ROWS = 2 * 2048;             // M*T
    uint16_t* qbuf = (uint16_t*)d_ws;
    uint16_t* kbuf = qbuf + ROWS * 1024;
    uint16_t* vbuf = kbuf + ROWS * 1024;
    uint16_t* cbuf = vbuf + ROWS * 1024;      // concat, bf16

    dim3 gg(1024 / GT, ROWS / GT);            // (8, 32)
    dim3 bb(256);

    gemm_bf16_wmma<true, false><<<gg, bb, 0, stream>>>(Q,  Wq, bq, qbuf, 1024);
    gemm_bf16_wmma<true, false><<<gg, bb, 0, stream>>>(Kp, Wk, bk, kbuf, 1024);
    gemm_bf16_wmma<true, false><<<gg, bb, 0, stream>>>(V,  Wv, bv, vbuf, 1024);

    attn_wmma<<<dim3(2048 / ABQ, 16, 2), bb, 0, stream>>>(qbuf, kbuf, vbuf, mask, cbuf);

    gemm_bf16_wmma<false, true><<<gg, bb, 0, stream>>>(cbuf, Wo, bo, out, 1024);
}
